// TopoGCNNRNS_84447646973974
// MI455X (gfx1250) — matrix operand order repaired and checked by reading, compiled, not run on hardware
//
#include <hip/hip_runtime.h>
#include <hip/hip_bf16.h>
#include <math.h>

// ---------------- problem constants (match reference) ----------------
static constexpr int NN  = 20000;
static constexpr int EE  = 160000;
static constexpr int CC  = 512;
static constexpr int DD  = 16;
static constexpr int HH  = 256;
static constexpr int KZ  = 2 * CC + DD;   // 1040
static constexpr int KZP = 1056;          // padded to multiple of 32

typedef __bf16 v16bf __attribute__((ext_vector_type(16)));
typedef float  v8f   __attribute__((ext_vector_type(8)));
union Frag16 { uint4 u[2]; v16bf v; };

__device__ __forceinline__ v8f wmma_bf16(v16bf a, v16bf b, v8f c) {
  return __builtin_amdgcn_wmma_f32_16x16x32_bf16(false, a, false, b, (short)0, c,
                                                 false, false);
}

// ---------------- small utility kernels ----------------
__global__ void k_zero(float* __restrict__ p, int n) {
  int i = blockIdx.x * 256 + threadIdx.x;
  if (i < n) p[i] = 0.0f;
}

__global__ void k_f2b(const float* __restrict__ a, __hip_bfloat16* __restrict__ b, int n) {
  int i = blockIdx.x * 256 + threadIdx.x;
  if (i < n) b[i] = __float2bfloat16(a[i]);
}

// goal_feat [N,512] fp32 -> zb[:, 256:768] bf16 (row stride 768)
__global__ void k_goal_pack(const float* __restrict__ g, __hip_bfloat16* __restrict__ zb, int n) {
  int i = blockIdx.x * 256 + threadIdx.x;
  if (i >= n) return;
  int node = i >> 9, c = i & 511;
  zb[(size_t)node * 768 + 256 + c] = __float2bfloat16(g[i]);
}

// ---------------- weight pack: fp32 [K x Ncols] -> WMMA B fragments ----------------
// Fragment (kt, nt): 32 lanes x 16 halfwords contiguous.
// lane L holds column n = nt*16 + (L&15); halfword h holds k = kt*32 + (L<16?0:8) + (h&7) + (h>=8?16:0).
__global__ void k_pack_w(const float* __restrict__ W, int K, int Ncols,
                         __hip_bfloat16* __restrict__ out, int ntBase, int NTtot, int total) {
  int idx = blockIdx.x * 256 + threadIdx.x;
  if (idx >= total) return;
  int h    = idx & 15;
  int lane = (idx >> 4) & 31;
  int t2   = idx >> 9;
  int ntc  = Ncols >> 4;
  int ntL  = t2 % ntc;
  int kt   = t2 / ntc;
  int k  = kt * 32 + ((lane >> 4) << 3) + (h & 7) + ((h >> 3) << 4);
  int nc = ntL * 16 + (lane & 15);
  float v = (k < K) ? W[(size_t)k * Ncols + nc] : 0.0f;
  out[((size_t)(kt * NTtot + ntBase + ntL)) * 512 + (size_t)lane * 16 + h] = __float2bfloat16(v);
}

// load a wave's 4 gate + 4 softplus B fragments for one K-step
__device__ __forceinline__ void load_bfrags(const uint4* __restrict__ Bv, int kt, int wave,
                                            int lane, Frag16 bF[4], Frag16 bS[4]) {
#pragma unroll
  for (int j = 0; j < 4; ++j) {
    int ntF = wave * 4 + j, ntS = 32 + wave * 4 + j;
    size_t fo = ((size_t)(kt * 64 + ntF)) * 64 + (size_t)lane * 2;
    size_t so = ((size_t)(kt * 64 + ntS)) * 64 + (size_t)lane * 2;
    bF[j].u[0] = Bv[fo]; bF[j].u[1] = Bv[fo + 1];
    bS[j].u[0] = Bv[so]; bS[j].u[1] = Bv[so + 1];
  }
}

// ---------------- CGConv edge kernel (the hot loop) ----------------
// 64 edges per block (4 M-tiles of 16) so each B fragment feeds 4 WMMAs (4x less
// L2 weight traffic). A fragments are gathered straight from global: each lane
// reads two contiguous 16B chunks of its edge's row; K-slices of 32 never cross
// the [x_dst | x_src | e] segment boundaries (512/1024 are 32-aligned).
// Wave w owns gate-ntiles w*4..w*4+3 and softplus-ntiles 32+w*4..32+w*4+3 so
// m = sigmoid(f)*softplus(s) fuses in registers; fp32 atomics into agg.
__global__ void __launch_bounds__(256)
k_cg_edge(const __hip_bfloat16* __restrict__ xb, const __hip_bfloat16* __restrict__ eab,
          const int* __restrict__ src, const int* __restrict__ dst,
          const __hip_bfloat16* __restrict__ Bp,
          const float* __restrict__ biasF, const float* __restrict__ biasS,
          float* __restrict__ agg) {
  __shared__ int dstS[64];
  const int tid = threadIdx.x;
  const int e0  = blockIdx.x * 64;
  if (tid < 64) dstS[tid] = dst[e0 + tid];
  __syncthreads();

  const int lane = tid & 31, wave = tid >> 5;
  const int m = lane & 15, grp = lane >> 4;

  // per-lane 32-bit element offsets of this lane's row for each M-tile
  unsigned dOff[4], sOff[4], eOff[4];
#pragma unroll
  for (int t = 0; t < 4; ++t) {
    int e = e0 + t * 16 + m;
    dOff[t] = (unsigned)dst[e] * CC;
    sOff[t] = (unsigned)src[e] * CC;
    eOff[t] = (unsigned)e * DD;
  }

  const v8f vzero = {0.f, 0.f, 0.f, 0.f, 0.f, 0.f, 0.f, 0.f};
  v8f accF[4][4], accS[4][4];
#pragma unroll
  for (int t = 0; t < 4; ++t)
#pragma unroll
    for (int j = 0; j < 4; ++j) { accF[t][j] = vzero; accS[t][j] = vzero; }

  const uint4* Bv = (const uint4*)Bp;

  // ---- K 0..511: x_dst segment ----
  for (int kt = 0; kt < 16; ++kt) {
    Frag16 bF[4], bS[4];
    load_bfrags(Bv, kt, wave, lane, bF, bS);
    const int off = kt * 32 + grp * 8;
#pragma unroll
    for (int t = 0; t < 4; ++t) {
      Frag16 a;
      a.u[0] = *(const uint4*)(xb + dOff[t] + off);
      a.u[1] = *(const uint4*)(xb + dOff[t] + off + 16);
#pragma unroll
      for (int j = 0; j < 4; ++j) {
        accF[t][j] = wmma_bf16(a.v, bF[j].v, accF[t][j]);
        accS[t][j] = wmma_bf16(a.v, bS[j].v, accS[t][j]);
      }
    }
  }
  // ---- K 512..1023: x_src segment ----
  for (int kt = 0; kt < 16; ++kt) {
    Frag16 bF[4], bS[4];
    load_bfrags(Bv, kt + 16, wave, lane, bF, bS);
    const int off = kt * 32 + grp * 8;
#pragma unroll
    for (int t = 0; t < 4; ++t) {
      Frag16 a;
      a.u[0] = *(const uint4*)(xb + sOff[t] + off);
      a.u[1] = *(const uint4*)(xb + sOff[t] + off + 16);
#pragma unroll
      for (int j = 0; j < 4; ++j) {
        accF[t][j] = wmma_bf16(a.v, bF[j].v, accF[t][j]);
        accS[t][j] = wmma_bf16(a.v, bS[j].v, accS[t][j]);
      }
    }
  }
  // ---- K 1024..1055: edge_attr (16) + zero pad (16) ----
  {
    Frag16 bF[4], bS[4];
    load_bfrags(Bv, 32, wave, lane, bF, bS);
#pragma unroll
    for (int t = 0; t < 4; ++t) {
      Frag16 a;
      a.u[0] = *(const uint4*)(eab + eOff[t] + grp * 8);
      a.u[1] = make_uint4(0u, 0u, 0u, 0u);
#pragma unroll
      for (int j = 0; j < 4; ++j) {
        accF[t][j] = wmma_bf16(a.v, bF[j].v, accF[t][j]);
        accS[t][j] = wmma_bf16(a.v, bS[j].v, accS[t][j]);
      }
    }
  }

  // epilogue: D layout — VGPR r: lanes 0-15 row r, lanes 16-31 row r+8; col = lane&15
#pragma unroll
  for (int t = 0; t < 4; ++t) {
#pragma unroll
    for (int j = 0; j < 4; ++j) {
      int col = (wave * 4 + j) * 16 + m;
      float fb = biasF[col], sb = biasS[col];
#pragma unroll
      for (int r = 0; r < 8; ++r) {
        int row = t * 16 + r + grp * 8;
        float f = accF[t][j][r] + fb;
        float s = accS[t][j][r] + sb;
        float sig = 1.0f / (1.0f + __expf(-f));
        float sp  = (s > 20.0f) ? s : log1pf(__expf(s));
        atomicAdd(&agg[(size_t)dstS[row] * CC + col], sig * sp);
      }
    }
  }
}

// ---------------- generic WMMA GEMM: C[M x Nout] = A(bf16) * Bpack (+bias, relu) ----------------
// One 16-row tile per block; 8 waves; wave owns NT column tiles; Nout must be NT*128.
template <int NT>
__global__ void __launch_bounds__(256)
k_gemm(const __hip_bfloat16* __restrict__ A, const __hip_bfloat16* __restrict__ Bp,
       const float* __restrict__ bias, float* __restrict__ C, int KP, int Nout, int relu) {
  __shared__ __hip_bfloat16 At[16 * 768];
  const int tid = threadIdx.x;
  const int rowBase = blockIdx.x * 16;
  const int cpr = KP >> 3;                 // uint4 chunks per row
  uint4* Atv = (uint4*)At;
  const uint4* Av = (const uint4*)(A + (size_t)rowBase * KP);
  for (int i = tid; i < 16 * cpr; i += 256) Atv[i] = Av[i];
  __syncthreads();

  const int lane = tid & 31, wave = tid >> 5;
  const int m = lane & 15, grp = lane >> 4;
  const int NTtot = Nout >> 4;
  const v8f vzero = {0.f, 0.f, 0.f, 0.f, 0.f, 0.f, 0.f, 0.f};
  v8f acc[NT];
#pragma unroll
  for (int j = 0; j < NT; ++j) acc[j] = vzero;

  const uint4* Bv = (const uint4*)Bp;
  for (int kt = 0; kt < (KP >> 5); ++kt) {
    Frag16 a;
    size_t ab = (size_t)m * cpr + (size_t)kt * 4 + grp;
    a.u[0] = Atv[ab];
    a.u[1] = Atv[ab + 2];
    Frag16 b[NT];
#pragma unroll
    for (int j = 0; j < NT; ++j) {
      size_t fo = ((size_t)(kt * NTtot + wave * NT + j)) * 64 + (size_t)lane * 2;
      b[j].u[0] = Bv[fo]; b[j].u[1] = Bv[fo + 1];
    }
#pragma unroll
    for (int j = 0; j < NT; ++j) acc[j] = wmma_bf16(a.v, b[j].v, acc[j]);
  }

#pragma unroll
  for (int j = 0; j < NT; ++j) {
    int ncol = (wave * NT + j) * 16 + m;
    float bv = bias ? bias[ncol] : 0.0f;
#pragma unroll
    for (int r = 0; r < 8; ++r) {
      int row = rowBase + r + grp * 8;
      float v = acc[j][r] + bv;
      if (relu) v = fmaxf(v, 0.0f);
      C[(size_t)row * Nout + ncol] = v;
    }
  }
}

// ---------------- CGConv node epilogue: relu(agg*BNscale + beta + x) ----------------
__global__ void k_cg_node(const float* __restrict__ agg, const float* __restrict__ g,
                          const float* __restrict__ be, const float* __restrict__ xres,
                          float* __restrict__ fOut, __hip_bfloat16* __restrict__ bOut, int n) {
  int i = blockIdx.x * 256 + threadIdx.x;
  if (i >= n) return;
  int c = i & (CC - 1);
  float scale = g[c] * rsqrtf(1.0f + 1e-5f);
  float v = fmaxf(agg[i] * scale + be[c] + xres[i], 0.0f);
  if (fOut) fOut[i] = v;
  bOut[i] = __float2bfloat16(v);
}

// ---------------- GCN helpers ----------------
__global__ void k_deg_count(const int* __restrict__ dst, float* __restrict__ deg, int n) {
  int i = blockIdx.x * 256 + threadIdx.x;
  if (i < n) atomicAdd(&deg[dst[i]], 1.0f);
}
__global__ void k_deg_fin(float* __restrict__ d, int n) {
  int i = blockIdx.x * 256 + threadIdx.x;
  if (i < n) d[i] = rsqrtf(d[i] + 1.0f);   // +1 self loop; deg>=1
}
// scatter: acc[dst] += dinv[s]*dinv[d] * hW[src];  4 channels per thread
__global__ void k_gcn_scatter(const int* __restrict__ src, const int* __restrict__ dst,
                              const float* __restrict__ dinv, const float* __restrict__ hW,
                              float* __restrict__ acc, int total) {
  int idx = blockIdx.x * 256 + threadIdx.x;
  if (idx >= total) return;
  int e = idx >> 6, q = idx & 63;
  int s = src[e], d = dst[e];
  float w = dinv[s] * dinv[d];
  float4 v = ((const float4*)(hW + (size_t)s * HH))[q];
  float* o = acc + (size_t)d * HH + q * 4;
  atomicAdd(o + 0, w * v.x);
  atomicAdd(o + 1, w * v.y);
  atomicAdd(o + 2, w * v.z);
  atomicAdd(o + 3, w * v.w);
}
// finalize: relu(acc + dinv^2*hW(self loop) + b) -> bf16 (stride bStride) and optional fp32
__global__ void k_gcn_fin(const float* __restrict__ acc, const float* __restrict__ hW,
                          const float* __restrict__ dinv, const float* __restrict__ b,
                          __hip_bfloat16* __restrict__ bOut, int bStride,
                          float* __restrict__ fOut, int n) {
  int i = blockIdx.x * 256 + threadIdx.x;
  if (i >= n) return;
  int node = i >> 8, c = i & (HH - 1);
  float di = dinv[node];
  float v = fmaxf(acc[i] + di * di * hW[i] + b[c], 0.0f);
  bOut[(size_t)node * bStride + c] = __float2bfloat16(v);
  if (fOut) fOut[i] = v;
}

// ---------------- final 512 -> 1 projection ----------------
__global__ void k_pred(const float* __restrict__ a, const float* __restrict__ w2,
                       const float* __restrict__ b2, float* __restrict__ out, int n) {
  int i = blockIdx.x * 256 + threadIdx.x;
  if (i >= n) return;
  const float4* av = (const float4*)(a + (size_t)i * CC);
  const float4* wv = (const float4*)w2;
  float s = 0.0f;
  for (int k = 0; k < CC / 4; ++k) {
    float4 x = av[k], y = wv[k];
    s += x.x * y.x + x.y * y.y + x.z * y.z + x.w * y.w;
  }
  out[i] = s + b2[0];
}

// ---------------- host driver ----------------
extern "C" void kernel_launch(void* const* d_in, const int* in_sizes, int n_in,
                              void* d_out, int out_size, void* d_ws, size_t ws_size,
                              hipStream_t stream) {
  (void)in_sizes; (void)n_in; (void)out_size;
  const float* x    = (const float*)d_in[0];
  const int*   ei   = (const int*)d_in[1];
  const int*   src  = ei;
  const int*   dst  = ei + EE;
  const float* ea   = (const float*)d_in[2];
  const float* goal = (const float*)d_in[3];
  const float *Wf1 = (const float*)d_in[4],  *bf1 = (const float*)d_in[5];
  const float *Ws1 = (const float*)d_in[6],  *bs1 = (const float*)d_in[7];
  const float *g1  = (const float*)d_in[8],  *be1 = (const float*)d_in[9];
  const float *Wf2 = (const float*)d_in[10], *bf2 = (const float*)d_in[11];
  const float *Ws2 = (const float*)d_in[12], *bs2 = (const float*)d_in[13];
  const float *g2  = (const float*)d_in[14], *be2 = (const float*)d_in[15];
  const float *W3  = (const float*)d_in[16], *b3  = (const float*)d_in[17];
  const float *W4  = (const float*)d_in[18], *b4  = (const float*)d_in[19];
  const float *Wd1 = (const float*)d_in[20], *bd1 = (const float*)d_in[21];
  const float *Wd2 = (const float*)d_in[22], *bd2 = (const float*)d_in[23];
  float* dout = (float*)d_out;

  // ---- workspace layout ----
  char* W = (char*)d_ws;
  size_t off = 0;
  auto take = [&](size_t bytes) { char* p = W + off; off += (bytes + 255) & ~size_t(255); return p; };
  __hip_bfloat16* Wc1p = (__hip_bfloat16*)take((size_t)KZP * 1024 * 2);
  __hip_bfloat16* Wc2p = (__hip_bfloat16*)take((size_t)KZP * 1024 * 2);
  __hip_bfloat16* W3p  = (__hip_bfloat16*)take((size_t)CC * HH * 2);
  __hip_bfloat16* W4p  = (__hip_bfloat16*)take((size_t)HH * HH * 2);
  __hip_bfloat16* Wd1p = (__hip_bfloat16*)take((size_t)(HH + CC) * CC * 2);
  __hip_bfloat16* eab  = (__hip_bfloat16*)take((size_t)EE * DD * 2);
  __hip_bfloat16* xb   = (__hip_bfloat16*)take((size_t)NN * CC * 2);
  float* h1f    = (float*)take((size_t)NN * CC * 4);
  float* agg    = (float*)take((size_t)NN * CC * 4);
  float* hW     = (float*)take((size_t)NN * HH * 4);
  float* acc256 = (float*)take((size_t)NN * HH * 4);
  __hip_bfloat16* hb256 = (__hip_bfloat16*)take((size_t)NN * HH * 2);
  __hip_bfloat16* zb    = (__hip_bfloat16*)take((size_t)NN * 768 * 2);
  float* mlpA   = (float*)take((size_t)NN * CC * 4);
  float* dinv   = (float*)take((size_t)NN * 4);
  if (off > ws_size) return;  // workspace too small — deterministic no-op

  auto B = [](int n) { return (n + 255) / 256; };

  // ---- input conversions + weight packing ----
  k_f2b<<<B(NN * CC), 256, 0, stream>>>(x, xb, NN * CC);
  k_f2b<<<B(EE * DD), 256, 0, stream>>>(ea, eab, EE * DD);
  {
    int tCG = 33 * (CC / 16) * 512;           // KT=33, 32 ntiles per half
    k_pack_w<<<B(tCG), 256, 0, stream>>>(Wf1, KZ, CC, Wc1p, 0,  64, tCG);
    k_pack_w<<<B(tCG), 256, 0, stream>>>(Ws1, KZ, CC, Wc1p, 32, 64, tCG);
    k_pack_w<<<B(tCG), 256, 0, stream>>>(Wf2, KZ, CC, Wc2p, 0,  64, tCG);
    k_pack_w<<<B(tCG), 256, 0, stream>>>(Ws2, KZ, CC, Wc2p, 32, 64, tCG);
    int t3 = (CC / 32) * (HH / 16) * 512;
    k_pack_w<<<B(t3), 256, 0, stream>>>(W3, CC, HH, W3p, 0, HH / 16, t3);
    int t4 = (HH / 32) * (HH / 16) * 512;
    k_pack_w<<<B(t4), 256, 0, stream>>>(W4, HH, HH, W4p, 0, HH / 16, t4);
    int t5 = ((HH + CC) / 32) * (CC / 16) * 512;
    k_pack_w<<<B(t5), 256, 0, stream>>>(Wd1, HH + CC, CC, Wd1p, 0, CC / 16, t5);
  }
  // degree normalization (graph-constant)
  k_zero<<<B(NN), 256, 0, stream>>>(dinv, NN);
  k_deg_count<<<B(EE), 256, 0, stream>>>(dst, dinv, EE);
  k_deg_fin<<<B(NN), 256, 0, stream>>>(dinv, NN);

  // ---- CGConv layer 1 ----
  k_zero<<<B(NN * CC), 256, 0, stream>>>(agg, NN * CC);
  k_cg_edge<<<EE / 64, 256, 0, stream>>>(xb, eab, src, dst, Wc1p, bf1, bs1, agg);
  k_cg_node<<<B(NN * CC), 256, 0, stream>>>(agg, g1, be1, x, h1f, xb, NN * CC);

  // ---- CGConv layer 2 (reads xb=h1 bf16, residual h1f fp32; writes h2 -> xb) ----
  k_zero<<<B(NN * CC), 256, 0, stream>>>(agg, NN * CC);
  k_cg_edge<<<EE / 64, 256, 0, stream>>>(xb, eab, src, dst, Wc2p, bf2, bs2, agg);
  k_cg_node<<<B(NN * CC), 256, 0, stream>>>(agg, g2, be2, h1f, nullptr, xb, NN * CC);

  // ---- GCN layer 3: hW = h2 @ W3; scatter; relu(+b3) -> hb256 ----
  k_gemm<2><<<NN / 16, 256, 0, stream>>>(xb, W3p, nullptr, hW, CC, HH, 0);
  k_zero<<<B(NN * HH), 256, 0, stream>>>(acc256, NN * HH);
  k_gcn_scatter<<<B(EE * 64), 256, 0, stream>>>(src, dst, dinv, hW, acc256, EE * 64);
  k_gcn_fin<<<B(NN * HH), 256, 0, stream>>>(acc256, hW, dinv, b3, hb256, HH, nullptr, NN * HH);

  // ---- GCN layer 4: h = relu(...) -> d_out[N..] and zb[:, :256] ----
  k_gemm<2><<<NN / 16, 256, 0, stream>>>(hb256, W4p, nullptr, hW, HH, HH, 0);
  k_zero<<<B(NN * HH), 256, 0, stream>>>(acc256, NN * HH);
  k_gcn_scatter<<<B(EE * 64), 256, 0, stream>>>(src, dst, dinv, hW, acc256, EE * 64);
  k_gcn_fin<<<B(NN * HH), 256, 0, stream>>>(acc256, hW, dinv, b4, zb, 768, dout + NN, NN * HH);

  // ---- distance MLP ----
  k_goal_pack<<<B(NN * CC), 256, 0, stream>>>(goal, zb, NN * CC);
  k_gemm<4><<<NN / 16, 256, 0, stream>>>(zb, Wd1p, bd1, mlpA, HH + CC, CC, 1);
  k_pred<<<B(NN), 256, 0, stream>>>(mlpA, Wd2, bd2, dout, NN);
}
